// PointEncoder_5566277616512
// MI455X (gfx1250) — matrix operand order repaired and checked by reading, not compiled, not run on hardware
//
#include <hip/hip_runtime.h>
#include <hip/hip_bf16.h>

#define N_ATOMS 20000
#define N_EDGES 320000
#define N_MOL   500
#define DIM     256
#define MDIM    128
#define NHEAD   4
#define DHEAD   32
#define NRBF    32
#define NLAYER  4
#define EMBD    32
#define OUTD    512
#define CUTOFF  6.0f
#define PI_F    3.14159265358979323846f

typedef __attribute__((ext_vector_type(16))) __bf16 v16bf;
typedef __attribute__((ext_vector_type(8)))  float v8f;

// ---------------- helpers ----------------

__device__ __forceinline__ __bf16 tobf(float x) { return (__bf16)x; }

__device__ __forceinline__ unsigned fkey(float x) {
    unsigned b = __float_as_uint(x);
    return (b & 0x80000000u) ? ~b : (b | 0x80000000u); // monotonic order-preserving key
}
__device__ __forceinline__ float keyToFloat(unsigned k) {
    unsigned b = (k & 0x80000000u) ? (k & 0x7FFFFFFFu) : ~k;
    return __uint_as_float(b);
}

__device__ __forceinline__ float gelu_tanh(float x) {
    float x3 = x * x * x;
    return 0.5f * x * (1.0f + tanhf(0.7978845608028654f * (x + 0.044715f * x3)));
}

union BfFrag { uint4 q[2]; v16bf v; };

// ---------------- zero ----------------

__global__ void k_zero(unsigned* __restrict__ p, long long n) {
    long long i = (long long)blockIdx.x * blockDim.x + threadIdx.x;
    if (i < n) p[i] = 0u;
}

// ---------------- weight pre-pack: f32 [K,N] -> bf16 WMMA-B fragment order ----
// fragment (nt16, kt): 32 lanes x 16 contiguous bf16; lane: half=lane>>4, col=nt16*16+(lane&15)
// element i of lane -> B[(kt*32 + half*8 + i + (i>=8 ? 8 : 0))*N + col]

__global__ void k_pack_b(const float* __restrict__ B, __bf16* __restrict__ P,
                         int K, int N) {
    int t = blockIdx.x * blockDim.x + threadIdx.x;   // one thread per (frag,lane)
    int Kt = K >> 5;
    int total = Kt * (N >> 4) * 32;
    if (t >= total) return;
    int lane = t & 31;
    int frag = t >> 5;
    int kt = frag % Kt;
    int nt16 = frag / Kt;
    int half = lane >> 4;
    int col = nt16 * 16 + (lane & 15);
    int kbase = kt * 32 + half * 8;
    __bf16* o = P + t * 16;
#pragma unroll
    for (int i = 0; i < 16; ++i) o[i] = tobf(B[(kbase + i + ((i >> 3) << 3)) * N + col]);
}

// ---------------- node init: h = [emb[z], pos] @ W_in + b_in ----------------

__global__ void k_init_h(const float* __restrict__ pos, const int* __restrict__ z,
                         const float* __restrict__ emb, const float* __restrict__ Win,
                         const float* __restrict__ bin, float* __restrict__ h) {
    int n = blockIdx.x, d = threadIdx.x;
    int zt = z[n];
    float acc = bin[d];
    const float* er = emb + zt * EMBD;
#pragma unroll 8
    for (int k = 0; k < EMBD; ++k) acc += er[k] * Win[k * DIM + d];
#pragma unroll
    for (int k = 0; k < 3; ++k) acc += pos[n * 3 + k] * Win[(EMBD + k) * DIM + d];
    h[n * DIM + d] = acc;
}

// ---------------- edge geometry + Bessel RBF ----------------

__global__ void k_edge_geom(const float* __restrict__ pos, const int* __restrict__ src,
                            const int* __restrict__ dst, float* __restrict__ unitv,
                            float* __restrict__ rbf) {
    int e = blockIdx.x * blockDim.x + threadIdx.x;
    if (e >= N_EDGES) return;
    int s = src[e], d = dst[e];
    float rx = pos[s * 3 + 0] - pos[d * 3 + 0];
    float ry = pos[s * 3 + 1] - pos[d * 3 + 1];
    float rz = pos[s * 3 + 2] - pos[d * 3 + 2];
    float dist = sqrtf(rx * rx + ry * ry + rz * rz + 1e-12f);
    float inv = 1.0f / dist;
    unitv[e * 3 + 0] = rx * inv;
    unitv[e * 3 + 1] = ry * inv;
    unitv[e * 3 + 2] = rz * inv;
    float env = (dist < CUTOFF) ? 0.5f * (cosf(PI_F * dist / CUTOFF) + 1.0f) : 0.0f;
    float sc = inv * env;
#pragma unroll 8
    for (int r = 0; r < NRBF; ++r) {
        float freq = (float)(r + 1) * (PI_F / CUTOFF);
        rbf[e * NRBF + r] = sinf(dist * freq) * sc;
    }
}

// ---------------- bf16 WMMA GEMM: C = act(A@B + bias) -------------------------
// one wave computes a 16x32 output tile; A f32 row-major [Mr,Kd]; B pre-packed bf16
// grid.x = Nc/32, grid.y = ceil(Mr/16)

template<int ACT, bool BIAS>
__global__ void k_wmma_gemm(const float* __restrict__ A, int lda,
                            const __bf16* __restrict__ Bp,
                            const float* __restrict__ bias,
                            float* __restrict__ C, int ldc,
                            int Mr, int Kd) {
    int lane = threadIdx.x & 31;
    int half = lane >> 4;
    int l15  = lane & 15;
    int row0 = blockIdx.y * 16;
    int rows_ok = Mr - row0;
    int arow = (l15 < rows_ok) ? l15 : 0;      // clamp once: branch-free K loop
    const float* Arow = A + (row0 + arow) * lda + (half << 3);
    int Kt = Kd >> 5;
    int nt0 = blockIdx.x * 2;                  // two 16-col fragments
    const __bf16* bp0 = Bp + (nt0 * Kt) * 512 + lane * 16;
    const __bf16* bp1 = bp0 + Kt * 512;

    v8f acc0 = {0.f, 0.f, 0.f, 0.f, 0.f, 0.f, 0.f, 0.f};
    v8f acc1 = {0.f, 0.f, 0.f, 0.f, 0.f, 0.f, 0.f, 0.f};

    for (int kt = 0; kt < Kt; ++kt) {
        int k0 = kt << 5;
        float4 a0 = *(const float4*)(Arow + k0);
        float4 a1 = *(const float4*)(Arow + k0 + 4);
        float4 a2 = *(const float4*)(Arow + k0 + 16);
        float4 a3 = *(const float4*)(Arow + k0 + 20);
        v16bf av;
        av[0]  = tobf(a0.x); av[1]  = tobf(a0.y); av[2]  = tobf(a0.z); av[3]  = tobf(a0.w);
        av[4]  = tobf(a1.x); av[5]  = tobf(a1.y); av[6]  = tobf(a1.z); av[7]  = tobf(a1.w);
        av[8]  = tobf(a2.x); av[9]  = tobf(a2.y); av[10] = tobf(a2.z); av[11] = tobf(a2.w);
        av[12] = tobf(a3.x); av[13] = tobf(a3.y); av[14] = tobf(a3.z); av[15] = tobf(a3.w);

        BfFrag b0, b1;
        b0.q[0] = *(const uint4*)(bp0 + kt * 512);
        b0.q[1] = *(const uint4*)(bp0 + kt * 512 + 8);
        b1.q[0] = *(const uint4*)(bp1 + kt * 512);
        b1.q[1] = *(const uint4*)(bp1 + kt * 512 + 8);

        acc0 = __builtin_amdgcn_wmma_f32_16x16x32_bf16(false, av, false, b0.v,
                                                       (short)0, acc0, false, false);
        acc1 = __builtin_amdgcn_wmma_f32_16x16x32_bf16(false, av, false, b1.v,
                                                       (short)0, acc1, false, false);
    }

    int n = blockIdx.x * 32 + l15;
    float bb0 = 0.f, bb1 = 0.f;
    if (BIAS) { bb0 = bias[n]; bb1 = bias[n + 16]; }   // hoisted, loop-invariant
    int mbase = row0 + (half << 3);
    if (rows_ok >= 16) {
#pragma unroll
        for (int r = 0; r < 8; ++r) {
            float v0 = acc0[r] + bb0;
            float v1 = acc1[r] + bb1;
            if (ACT == 1) { v0 = gelu_tanh(v0); v1 = gelu_tanh(v1); }
            C[(mbase + r) * ldc + n] = v0;
            C[(mbase + r) * ldc + n + 16] = v1;
        }
    } else {
#pragma unroll
        for (int r = 0; r < 8; ++r) {
            if (mbase + r < Mr) {
                float v0 = acc0[r] + bb0;
                float v1 = acc1[r] + bb1;
                if (ACT == 1) { v0 = gelu_tanh(v0); v1 = gelu_tanh(v1); }
                C[(mbase + r) * ldc + n] = v0;
                C[(mbase + r) * ldc + n + 16] = v1;
            }
        }
    }
}

// ---------------- attention logits + segment max ----------------

__global__ void k_edge_logits(const float* __restrict__ qkv, const int* __restrict__ src,
                              const int* __restrict__ dst, float* __restrict__ logits,
                              unsigned* __restrict__ attmax) {
    int t = blockIdx.x * blockDim.x + threadIdx.x;
    if (t >= N_EDGES * NHEAD) return;
    int e = t >> 2, hh = t & 3;
    int d = dst[e], s = src[e];
    const float* q = qkv + d * 384 + hh * DHEAD;          // Q at col 0
    const float* k = qkv + s * 384 + MDIM + hh * DHEAD;   // K at col 128
    float acc = 0.f;
#pragma unroll 8
    for (int i = 0; i < DHEAD; ++i) acc += q[i] * k[i];
    acc *= 0.17677669529663687f;   // 1/sqrt(32)
    logits[t] = acc;
    atomicMax(&attmax[d * NHEAD + hh], fkey(acc));
}

// ---------------- exp + segment sum ----------------

__global__ void k_edge_exp(const int* __restrict__ dst, float* __restrict__ logits,
                           const unsigned* __restrict__ attmax, float* __restrict__ attsum) {
    int t = blockIdx.x * blockDim.x + threadIdx.x;
    if (t >= N_EDGES * NHEAD) return;
    int e = t >> 2, hh = t & 3;
    int d = dst[e];
    float m = keyToFloat(attmax[d * NHEAD + hh]);
    float ev = expf(logits[t] - m);
    logits[t] = ev;
    atomicAdd(&attsum[d * NHEAD + hh], ev);
}

// ---------------- fused rbf filter * value * attention -> me(bf16), msg ----------------

__global__ void k_edge_me(const float* __restrict__ qkv, const int* __restrict__ src,
                          const int* __restrict__ dst, const float* __restrict__ rbf,
                          const float* __restrict__ Wrbf_l, const float* __restrict__ logits,
                          const float* __restrict__ attsum, __bf16* __restrict__ me,
                          float* __restrict__ msg) {
    int e = blockIdx.x, m = threadIdx.x;   // block = 128
    int s = src[e], d = dst[e];
    const float* rb = rbf + e * NRBF;
    float ew = 0.f;
#pragma unroll 8
    for (int r = 0; r < NRBF; ++r) ew += rb[r] * Wrbf_l[r * MDIM + m];
    int hh = m >> 5;
    float ev = logits[e * NHEAD + hh];
    float a = ev / (attsum[d * NHEAD + hh] + 1e-12f);
    float vv = qkv[s * 384 + 256 + m] * ew;   // V at col 256
    float mev = a * vv;
    me[e * MDIM + m] = tobf(mev);
    atomicAdd(&msg[d * MDIM + m], mev);
}

// ---------------- WMMA gate GEMM fused with vfeat scatter ----------------
// 16 edges x 32 cols of gate = me[16x128](bf16) @ packed W_vgate
// grid.x = E/16, grid.y = 256/32

__global__ void k_gate_scatter(const __bf16* __restrict__ me, const __bf16* __restrict__ Wgp,
                               const int* __restrict__ dst, const float* __restrict__ unitv,
                               float* __restrict__ vfeat) {
    int lane = threadIdx.x & 31;
    int half = lane >> 4;
    int l15  = lane & 15;
    int et = blockIdx.x;
    int n0 = blockIdx.y * 32;
    const int Kt = MDIM >> 5;   // 4
    const __bf16* Arow = me + (et * 16 + l15) * MDIM + (half << 3);
    const __bf16* bp0 = Wgp + (blockIdx.y * 2 * Kt) * 512 + lane * 16;
    const __bf16* bp1 = bp0 + Kt * 512;

    v8f acc0 = {0.f, 0.f, 0.f, 0.f, 0.f, 0.f, 0.f, 0.f};
    v8f acc1 = {0.f, 0.f, 0.f, 0.f, 0.f, 0.f, 0.f, 0.f};

#pragma unroll
    for (int kt = 0; kt < Kt; ++kt) {
        int k0 = kt << 5;
        BfFrag ua, b0, b1;
        ua.q[0] = *(const uint4*)(Arow + k0);
        ua.q[1] = *(const uint4*)(Arow + k0 + 16);
        b0.q[0] = *(const uint4*)(bp0 + kt * 512);
        b0.q[1] = *(const uint4*)(bp0 + kt * 512 + 8);
        b1.q[0] = *(const uint4*)(bp1 + kt * 512);
        b1.q[1] = *(const uint4*)(bp1 + kt * 512 + 8);
        acc0 = __builtin_amdgcn_wmma_f32_16x16x32_bf16(false, ua.v, false, b0.v,
                                                       (short)0, acc0, false, false);
        acc1 = __builtin_amdgcn_wmma_f32_16x16x32_bf16(false, ua.v, false, b1.v,
                                                       (short)0, acc1, false, false);
    }

#pragma unroll
    for (int r = 0; r < 8; ++r) {
        int e = et * 16 + r + (half << 3);
        int dn = dst[e];
        float ux = unitv[e * 3 + 0], uy = unitv[e * 3 + 1], uz = unitv[e * 3 + 2];
        int n = n0 + l15;
        float g0 = acc0[r], g1 = acc1[r];
        float* vb = vfeat + dn * (3 * DIM) + n;
        atomicAdd(vb,                g0 * ux);
        atomicAdd(vb + DIM,          g0 * uy);
        atomicAdd(vb + 2 * DIM,      g0 * uz);
        atomicAdd(vb + 16,           g1 * ux);
        atomicAdd(vb + DIM + 16,     g1 * uy);
        atomicAdd(vb + 2 * DIM + 16, g1 * uz);
    }
}

// ---------------- vector norms ----------------

__global__ void k_vnorm(const float* __restrict__ vfeat, float* __restrict__ vnorm) {
    int n = blockIdx.x, d = threadIdx.x;
    const float* vb = vfeat + n * (3 * DIM) + d;
    float x = vb[0], y = vb[DIM], z = vb[2 * DIM];
    vnorm[n * DIM + d] = sqrtf(x * x + y * y + z * z + 1e-8f);
}

// ---------------- residual + layernorm * gamma (in-place on h) ----------------

__global__ void k_res_ln(float* __restrict__ h, const float* __restrict__ upd2,
                         const float* __restrict__ vs, const float* __restrict__ gamma_l) {
    __shared__ float red[DIM];
    int n = blockIdx.x, d = threadIdx.x;
    int idx = n * DIM + d;
    float x = h[idx] + upd2[idx] + vs[idx];
    red[d] = x; __syncthreads();
    for (int s = DIM / 2; s > 0; s >>= 1) {
        if (d < s) red[d] += red[d + s];
        __syncthreads();
    }
    float mu = red[0] * (1.0f / DIM);
    __syncthreads();
    float c = x - mu;
    red[d] = c * c; __syncthreads();
    for (int s = DIM / 2; s > 0; s >>= 1) {
        if (d < s) red[d] += red[d + s];
        __syncthreads();
    }
    float var = red[0] * (1.0f / DIM);
    h[idx] = c * rsqrtf(var + 1e-5f) * gamma_l[d];
}

// ---------------- graph softmax pooling ----------------

__global__ void k_node_att(const float* __restrict__ h, const float* __restrict__ watt,
                           const int* __restrict__ batch, float* __restrict__ nodeS,
                           unsigned* __restrict__ gmax) {
    __shared__ float red[DIM];
    int n = blockIdx.x, d = threadIdx.x;
    red[d] = h[n * DIM + d] * watt[d];
    __syncthreads();
    for (int s = DIM / 2; s > 0; s >>= 1) {
        if (d < s) red[d] += red[d + s];
        __syncthreads();
    }
    if (d == 0) {
        float v = red[0];
        nodeS[n] = v;
        atomicMax(&gmax[batch[n]], fkey(v));
    }
}

__global__ void k_node_exp(const float* __restrict__ nodeS, const int* __restrict__ batch,
                           const unsigned* __restrict__ gmax, float* __restrict__ nodeE,
                           float* __restrict__ gsum) {
    int n = blockIdx.x * blockDim.x + threadIdx.x;
    if (n >= N_ATOMS) return;
    int b = batch[n];
    float ev = expf(nodeS[n] - keyToFloat(gmax[b]));
    nodeE[n] = ev;
    atomicAdd(&gsum[b], ev);
}

__global__ void k_node_pool(const float* __restrict__ h, const int* __restrict__ batch,
                            const float* __restrict__ nodeE, const float* __restrict__ gsum,
                            float* __restrict__ pooled) {
    int n = blockIdx.x, d = threadIdx.x;
    int b = batch[n];
    float a = nodeE[n] / (gsum[b] + 1e-12f);
    atomicAdd(&pooled[b * DIM + d], a * h[n * DIM + d]);
}

// ---------------- host orchestration ----------------

static inline void zero_buf(void* p, long long n_u32, hipStream_t s) {
    int grid = (int)((n_u32 + 255) / 256);
    k_zero<<<grid, 256, 0, s>>>((unsigned*)p, n_u32);
}

static inline void pack_b(const float* B, __bf16* P, int K, int N, hipStream_t s) {
    int total = (K / 32) * (N / 16) * 32;
    k_pack_b<<<(total + 255) / 256, 256, 0, s>>>(B, P, K, N);
}

extern "C" void kernel_launch(void* const* d_in, const int* in_sizes, int n_in,
                              void* d_out, int out_size, void* d_ws, size_t ws_size,
                              hipStream_t stream) {
    const float* pos    = (const float*)d_in[0];
    const int*   z      = (const int*)d_in[1];
    const int*   batch  = (const int*)d_in[2];
    const int*   eidx   = (const int*)d_in[3];
    const float* emb    = (const float*)d_in[4];
    const float* Win    = (const float*)d_in[5];
    const float* bin    = (const float*)d_in[6];
    const float* Wq     = (const float*)d_in[7];
    const float* Wk     = (const float*)d_in[8];
    const float* Wv     = (const float*)d_in[9];
    const float* Wrbf   = (const float*)d_in[10];
    const float* Wmsg   = (const float*)d_in[11];
    const float* Wvgate = (const float*)d_in[12];
    const float* Wvs    = (const float*)d_in[13];
    const float* Wnode  = (const float*)d_in[14];
    const float* bnode  = (const float*)d_in[15];
    const float* gamma  = (const float*)d_in[16];
    const float* watt   = (const float*)d_in[17];
    const float* Wout   = (const float*)d_in[18];
    const float* bout   = (const float*)d_in[19];

    const int* src = eidx;                 // edge_index[0]
    const int* dst = eidx + N_EDGES;       // edge_index[1]

    // workspace carve-out
    char* base = (char*)d_ws;
    size_t off = 0;
    auto alloc = [&](size_t bytes) -> char* {
        char* p = base + off;
        off += (bytes + 255) & ~(size_t)255;
        return p;
    };
    float*    h      = (float*)alloc((size_t)N_ATOMS * DIM * 4);
    float*    qkv    = (float*)alloc((size_t)N_ATOMS * 384 * 4);
    float*    rbf    = (float*)alloc((size_t)N_EDGES * NRBF * 4);
    float*    unitv  = (float*)alloc((size_t)N_EDGES * 3 * 4);
    float*    logits = (float*)alloc((size_t)N_EDGES * NHEAD * 4);
    unsigned* attmax = (unsigned*)alloc((size_t)N_ATOMS * NHEAD * 4);
    float*    attsum = (float*)alloc((size_t)N_ATOMS * NHEAD * 4);
    __bf16*   me     = (__bf16*)alloc((size_t)N_EDGES * MDIM * 2);
    float*    msg    = (float*)alloc((size_t)N_ATOMS * MDIM * 4);
    float*    vfeat  = (float*)alloc((size_t)N_ATOMS * 3 * DIM * 4);
    float*    upd1   = (float*)alloc((size_t)N_ATOMS * DIM * 4);
    float*    upd2   = (float*)alloc((size_t)N_ATOMS * DIM * 4);
    float*    vnorm  = (float*)alloc((size_t)N_ATOMS * DIM * 4);
    float*    vs     = (float*)alloc((size_t)N_ATOMS * DIM * 4);
    float*    nodeS  = (float*)alloc((size_t)N_ATOMS * 4);
    float*    nodeE  = (float*)alloc((size_t)N_ATOMS * 4);
    unsigned* gmax   = (unsigned*)alloc((size_t)N_MOL * 4);
    float*    gsum   = (float*)alloc((size_t)N_MOL * 4);
    float*    pooled = (float*)alloc((size_t)N_MOL * DIM * 4);
    // packed bf16 weights (fragment order), per layer
    // pQKV: fused [256 x 384] = Wq cols 0..127 | Wk 128..255 | Wv 256..383
    __bf16* pQKV[NLAYER]; __bf16* pMsg[NLAYER]; __bf16* pG[NLAYER];
    __bf16* pNode[NLAYER]; __bf16* pVs[NLAYER];
    for (int l = 0; l < NLAYER; ++l) {
        pQKV[l]  = (__bf16*)alloc((size_t)DIM * MDIM * 3 * 2);
        pMsg[l]  = (__bf16*)alloc((size_t)MDIM * DIM * 2);
        pG[l]    = (__bf16*)alloc((size_t)MDIM * DIM * 2);
        pNode[l] = (__bf16*)alloc((size_t)DIM * DIM * 2);
        pVs[l]   = (__bf16*)alloc((size_t)DIM * DIM * 2);
    }
    __bf16* pOut = (__bf16*)alloc((size_t)DIM * OUTD * 2);
    (void)ws_size; (void)n_in; (void)in_sizes; (void)out_size;

    // pack all weights to bf16 fragment order (once per launch).
    // fragment order is column-group-major, so packing Wq/Wk/Wv into adjacent
    // slices of pQKV yields a valid packed [256,384] matrix.
    for (int l = 0; l < NLAYER; ++l) {
        pack_b(Wq     + (size_t)l * DIM * MDIM, pQKV[l],                  DIM,  MDIM, stream);
        pack_b(Wk     + (size_t)l * DIM * MDIM, pQKV[l] + DIM * MDIM,     DIM,  MDIM, stream);
        pack_b(Wv     + (size_t)l * DIM * MDIM, pQKV[l] + 2 * DIM * MDIM, DIM,  MDIM, stream);
        pack_b(Wmsg   + (size_t)l * MDIM * DIM, pMsg[l],  MDIM, DIM,  stream);
        pack_b(Wvgate + (size_t)l * MDIM * DIM, pG[l],    MDIM, DIM,  stream);
        pack_b(Wnode  + (size_t)l * DIM * DIM,  pNode[l], DIM,  DIM,  stream);
        pack_b(Wvs    + (size_t)l * DIM * DIM,  pVs[l],   DIM,  DIM,  stream);
    }
    pack_b(Wout, pOut, DIM, OUTD, stream);

    // vfeat accumulates across layers -> zero once
    zero_buf(vfeat, (long long)N_ATOMS * 3 * DIM, stream);

    k_init_h<<<N_ATOMS, DIM, 0, stream>>>(pos, z, emb, Win, bin, h);
    k_edge_geom<<<(N_EDGES + 255) / 256, 256, 0, stream>>>(pos, src, dst, unitv, rbf);

    for (int l = 0; l < NLAYER; ++l) {
        const float* Wrbf_l  = Wrbf  + (size_t)l * NRBF * MDIM;
        const float* bnode_l = bnode + (size_t)l * DIM;
        const float* gamma_l = gamma + (size_t)l * DIM;

        zero_buf(attmax, (long long)N_ATOMS * NHEAD, stream);
        zero_buf(attsum, (long long)N_ATOMS * NHEAD, stream);
        zero_buf(msg,    (long long)N_ATOMS * MDIM, stream);

        // fused QKV projection (WMMA, packed-B, single pass over h)
        dim3 gq(384 / 32, N_ATOMS / 16);
        k_wmma_gemm<0, false><<<gq, 32, 0, stream>>>(h, DIM, pQKV[l], nullptr, qkv, 384, N_ATOMS, DIM);

        int eh = N_EDGES * NHEAD;
        k_edge_logits<<<(eh + 255) / 256, 256, 0, stream>>>(qkv, src, dst, logits, attmax);
        k_edge_exp<<<(eh + 255) / 256, 256, 0, stream>>>(dst, logits, attmax, attsum);
        k_edge_me<<<N_EDGES, MDIM, 0, stream>>>(qkv, src, dst, rbf, Wrbf_l, logits, attsum, me, msg);

        // gate GEMM + equivariant scatter (WMMA, packed-B)
        dim3 gg(N_EDGES / 16, DIM / 32);
        k_gate_scatter<<<gg, 32, 0, stream>>>(me, pG[l], dst, unitv, vfeat);

        // node MLPs (WMMA, packed-B)
        dim3 gn(DIM / 32, N_ATOMS / 16);
        k_wmma_gemm<1, false><<<gn, 32, 0, stream>>>(msg, MDIM, pMsg[l], nullptr, upd1, DIM, N_ATOMS, MDIM);
        k_wmma_gemm<1, true><<<gn, 32, 0, stream>>>(upd1, DIM, pNode[l], bnode_l, upd2, DIM, N_ATOMS, DIM);

        k_vnorm<<<N_ATOMS, DIM, 0, stream>>>(vfeat, vnorm);
        k_wmma_gemm<0, false><<<gn, 32, 0, stream>>>(vnorm, DIM, pVs[l], nullptr, vs, DIM, N_ATOMS, DIM);

        k_res_ln<<<N_ATOMS, DIM, 0, stream>>>(h, upd2, vs, gamma_l);
    }

    // graph softmax pooling
    zero_buf(gmax,   N_MOL, stream);
    zero_buf(gsum,   N_MOL, stream);
    zero_buf(pooled, (long long)N_MOL * DIM, stream);

    k_node_att<<<N_ATOMS, DIM, 0, stream>>>(h, watt, batch, nodeS, gmax);
    k_node_exp<<<(N_ATOMS + 255) / 256, 256, 0, stream>>>(nodeS, batch, gmax, nodeE, gsum);
    k_node_pool<<<N_ATOMS, DIM, 0, stream>>>(h, batch, nodeE, gsum, pooled);

    // output bottleneck (WMMA, packed-B, row-bounded M=500)
    dim3 go(OUTD / 32, (N_MOL + 15) / 16);
    k_wmma_gemm<0, true><<<go, 32, 0, stream>>>(pooled, DIM, pOut, bout, (float*)d_out, OUTD, N_MOL, DIM);
}
